// DNCClassifier_82635170775168
// MI455X (gfx1250) — compile-verified, matrix operand order checked by reading, not run on hardware
//
#include <hip/hip_runtime.h>
#include <hip/hip_bf16.h>

// ---------------------------------------------------------------------------
// DNC classifier forward for MI455X (gfx1250, wave32, WMMA + async-to-LDS).
//
//   Phase A (parallel kernels): repack weights into WMMA fragment order (f16),
//     precompute Gx[b,t,n] = x_t @ W_ih.T + b_ih + b_hh for all steps (f16).
//   Phase B (persistent kernel, 1 workgroup of 1024 threads = 32 waves):
//     per step: stage W_hh K-slices into LDS via global_load_async_to_lds_b128
//     (double buffered, overlapped with WMMA), gates = h @ W_hh.T via
//     v_wmma_f32_16x16x32_f16 in TWO half-passes (8 live accumulators each to
//     stay inside the 128-VGPR/wave budget), LSTM elementwise with c in
//     registers, xi = h @ W_xi.T via WMMA from LDS-staged W_xi, DNC memory
//     update (1 thread / batch element, state in L2-resident scratch).
// ---------------------------------------------------------------------------

typedef __attribute__((ext_vector_type(16))) _Float16 v16h;
typedef __attribute__((ext_vector_type(8)))  float    v8f;

#define BB   128
#define TT   512
#define INX  27
#define NNIN 47
#define HH   256
#define MM   10
#define WCC  20
#define OUTN 128
#define EPSF 1e-6f

// workspace layout (bytes)
#define OFF_WHH   0u                    // 8*64*32*16 halves = 512 KB
#define OFF_WXI   524288u               // 8*6*32*16 halves  = 48 KB
#define OFF_WFC   573440u               // 8*8*32*16 halves  = 64 KB
#define OFF_STATE 638976u               // 128 * 384 f32     = 192 KB
#define OFF_GX    835584u               // 128*512*1024 f16  = 128 MB

// LDS layout (dynamic shared, starts at LDS offset 0: no static __shared__)
#define LDS_HA      0u                  // 2 x 65536 B   : h fragments, dbl buf
#define LDS_BSTAGE  131072u             // 2 x 65536 B   : W_hh slice, dbl buf
#define LDS_XI      262144u             // 128*96*4 B    : xi
#define LDS_BYTES   311296u             // 304 KB total

__device__ __forceinline__ float sigm(float x) { return 1.0f / (1.0f + __expf(-x)); }
__device__ __forceinline__ float softplusf(float x) {
  return fmaxf(x, 0.0f) + log1pf(__expf(-fabsf(x)));
}

// gfx1250 async global->LDS copy (ASYNCcnt), per-lane 16 bytes.
__device__ __forceinline__ void async_b128(unsigned int lds_off, const _Float16* g) {
  asm volatile("global_load_async_to_lds_b128 %0, %1, off"
               :: "v"(lds_off), "v"(g)
               : "memory");
}
__device__ __forceinline__ void wait_async0() {
  asm volatile("s_wait_asynccnt 0x0" ::: "memory");
}

// ---------------------------------------------------------------------------
// Fragment packing (16-bit A/B tiles): lanes 0-15 hold K in {0..7,16..23},
// lanes 16-31 hold K in {8..15,24..31}; half index = (k&7) + 8*((k>>4)&1).
// ---------------------------------------------------------------------------

__global__ void pack_whh_kernel(const float* __restrict__ W_hh, _Float16* __restrict__ dst) {
  int idx = blockIdx.x * 256 + threadIdx.x;         // 0 .. 262143
  int hidx = idx & 15;
  int lane = (idx >> 4) & 31;
  int nt   = (idx >> 9) & 63;
  int kt   = idx >> 15;
  int klocal = (hidx & 7) + ((hidx >> 3) << 4) + ((lane >> 4) << 3);
  int k = kt * 32 + klocal;
  int n = nt * 16 + (lane & 15);
  dst[idx] = (_Float16)W_hh[n * HH + k];
}

__global__ void pack_wxi_kernel(const float* __restrict__ W_xi, _Float16* __restrict__ dst) {
  int idx = blockIdx.x * 256 + threadIdx.x;         // 0 .. 24575
  int hidx = idx & 15;
  int lane = (idx >> 4) & 31;
  int tile = idx >> 9;                              // kt*6 + nt
  int nt = tile % 6, kt = tile / 6;
  int klocal = (hidx & 7) + ((hidx >> 3) << 4) + ((lane >> 4) << 3);
  int k = kt * 32 + klocal;
  int n = nt * 16 + (lane & 15);
  dst[idx] = (n < 88) ? (_Float16)W_xi[n * HH + k] : (_Float16)0.0f;
}

__global__ void pack_wfc_kernel(const float* __restrict__ W_fc, _Float16* __restrict__ dst) {
  int idx = blockIdx.x * 256 + threadIdx.x;         // 0 .. 32767
  int hidx = idx & 15;
  int lane = (idx >> 4) & 31;
  int tile = idx >> 9;                              // kt*8 + nt
  int nt = tile & 7, kt = tile >> 3;
  int klocal = (hidx & 7) + ((hidx >> 3) << 4) + ((lane >> 4) << 3);
  int k = kt * 32 + klocal;
  int n = nt * 16 + (lane & 15);
  dst[idx] = (_Float16)W_fc[n * HH + k];
}

// Gx[b][t][n] = b_ih[n] + b_hh[n] + sum_{k<27} x[b,t,k] * W_ih[n,k]
__global__ void gx_kernel(const float* __restrict__ x, const float* __restrict__ W_ih,
                          const float* __restrict__ b_ih, const float* __restrict__ b_hh,
                          _Float16* __restrict__ gx) {
  long id = (long)blockIdx.x * 256 + threadIdx.x;
  int n = (int)(id & 1023);
  long bt = id >> 10;
  const float* xp = x + bt * INX;
  const float* wp = W_ih + (long)n * NNIN;
  float s = b_ih[n] + b_hh[n];
#pragma unroll
  for (int k = 0; k < INX; ++k) s += xp[k] * wp[k];
  gx[id] = (_Float16)s;
}

// ---------------------------------------------------------------------------
// persistent recurrence kernel: 1 block x 1024 threads (32 waves)
// ---------------------------------------------------------------------------
__global__ __launch_bounds__(1024)
void dnc_persistent_kernel(const _Float16* __restrict__ whh_p,
                           const _Float16* __restrict__ wxi_p,
                           const _Float16* __restrict__ wfc_p,
                           const _Float16* __restrict__ gx,
                           float* __restrict__ state,
                           const float* __restrict__ b_xi,
                           const float* __restrict__ b_fc,
                           float* __restrict__ out) {
  extern __shared__ char smem[];
  _Float16* hA  = (_Float16*)(smem + LDS_HA);       // 2 x 32768 halves
  _Float16* bsS = (_Float16*)(smem + LDS_BSTAGE);   // 2 x 32768 halves
  float*    xiL = (float*)(smem + LDS_XI);          // 128 x 96 f32

  const int tid  = threadIdx.x;
  const int wid  = tid >> 5;
  const int lane = tid & 31;
  const int mrow_off = (lane >= 16) ? 8 : 0;
  const int ncol = lane & 15;

  // wave tile assignment: one b-tile (mt), four h-col tiles (nb..nb+3)
  const int mt = wid >> 2;
  const int nb = (wid & 3) << 2;

  // ---- init: h0 = 0, c0 = 0, DNC state per reference reset ----
  for (int i = tid; i < 32768; i += 1024) hA[i] = (_Float16)0.0f;
  if (tid < BB) {
    float* st = state + tid * 384;
    for (int i = 0;   i < 200; ++i) st[i] = EPSF;   // mem
    for (int i = 200; i < 300; ++i) st[i] = 0.0f;   // link
    for (int i = 300; i < 310; ++i) st[i] = 0.0f;   // prec
    for (int i = 310; i < 320; ++i) st[i] = EPSF;   // rw
    for (int i = 320; i < 330; ++i) st[i] = EPSF;   // ww
    for (int i = 330; i < 340; ++i) st[i] = 0.0f;   // usage
  }
  float creg[4][8];
#pragma unroll
  for (int q = 0; q < 4; ++q)
#pragma unroll
    for (int r = 0; r < 8; ++r) creg[q][r] = 0.0f;
  __syncthreads();

#pragma unroll 1
  for (int t = 0; t < TT; ++t) {
    const int cur = t & 1, nxt = cur ^ 1;
    const _Float16* hCur = hA + cur * 32768;
    _Float16*       hNxt = hA + nxt * 32768;

    // ===== phase 1: gates = h @ W_hh.T, two half-passes (8 accs live) =====
#pragma unroll 1
    for (int pass = 0; pass < 2; ++pass) {
      v8f acc[2][4];
#pragma unroll
      for (int qq = 0; qq < 2; ++qq)
#pragma unroll
        for (int ch = 0; ch < 4; ++ch)
#pragma unroll
          for (int e = 0; e < 8; ++e) acc[qq][ch][e] = 0.0f;

      // prologue: stage kt=0 slice (64 KB) into buffer 0
      {
        unsigned int lb = LDS_BSTAGE + tid * 16;
        const _Float16* g = whh_p + tid * 8;
        async_b128(lb,          g);
        async_b128(lb + 16384,  g + 8192);
        async_b128(lb + 32768,  g + 16384);
        async_b128(lb + 49152,  g + 24576);
      }
#pragma unroll 1
      for (int kt = 0; kt < 8; ++kt) {
        wait_async0();            // this kt's slice (issued last iter) done
        __syncthreads();          // visible; prior readers of next buf done
        if (kt < 7) {             // stage kt+1 slice, overlapped with compute
          unsigned int lb = LDS_BSTAGE + (((kt + 1) & 1) << 16) + tid * 16;
          const _Float16* g = whh_p + (kt + 1) * 32768 + tid * 8;
          async_b128(lb,          g);
          async_b128(lb + 16384,  g + 8192);
          async_b128(lb + 32768,  g + 16384);
          async_b128(lb + 49152,  g + 24576);
        }
        const _Float16* bs = bsS + (kt & 1) * 32768;
        v16h a = *(const v16h*)(hCur + (((mt << 3) + kt) * 32 + lane) * 16);
#pragma unroll
        for (int qq = 0; qq < 2; ++qq) {
#pragma unroll
          for (int ch = 0; ch < 4; ++ch) {
            const int nt = (nb + (pass << 1) + qq) + (ch << 4);
            v16h bf = *(const v16h*)(bs + (nt * 32 + lane) * 16);
            acc[qq][ch] = __builtin_amdgcn_wmma_f32_16x16x32_f16(
                false, a, false, bf, (short)0, acc[qq][ch], false, false);
          }
        }
      }
      if (pass == 1) {
        // stage W_xi (48 KB) into buffer 0, overlapped with LSTM elementwise
        unsigned int lb = LDS_BSTAGE + tid * 16;
        const _Float16* g = wxi_p + tid * 8;
        async_b128(lb,          g);
        async_b128(lb + 16384,  g + 8192);
        async_b128(lb + 32768,  g + 16384);
      }
      // LSTM elementwise for this half: element (q, r, lane) -> (b, j)
#pragma unroll
      for (int qq = 0; qq < 2; ++qq) {
        const int q = (pass << 1) + qq;
#pragma unroll
        for (int r = 0; r < 8; ++r) {
          const int bi = (mt << 4) + r + mrow_off;
          const int j  = ((nb + q) << 4) + ncol;          // hidden col 0..255
          const _Float16* gxp = gx + ((long)bi * TT + t) * 1024;
          float gi = acc[qq][0][r] + (float)__builtin_nontemporal_load(gxp + j);
          float gf = acc[qq][1][r] + (float)__builtin_nontemporal_load(gxp + 256 + j);
          float gg = acc[qq][2][r] + (float)__builtin_nontemporal_load(gxp + 512 + j);
          float go = acc[qq][3][r] + (float)__builtin_nontemporal_load(gxp + 768 + j);
          float c  = sigm(gf) * creg[q][r] + sigm(gi) * tanhf(gg);
          float h  = sigm(go) * tanhf(c);                 // |h|<1 => clip no-op
          creg[q][r] = c;
          const int kt2 = j >> 5, kl = j & 31;
          const int wl   = (bi & 15) + (((kl >> 3) & 1) << 4);
          const int hidx = (kl & 7) + (((kl >> 4) & 1) << 3);
          hNxt[((((bi >> 4) << 3) + kt2) * 32 + wl) * 16 + hidx] = (_Float16)h;
        }
      }
    }
    wait_async0();
    __syncthreads();

    // ===== phase 2: xi = h_t @ W_xi.T (WMMA, 8x6 tiles, B from LDS) =====
#pragma unroll 1
    for (int rep = 0; rep < 2; ++rep) {
      const int tile = wid + (rep << 5);
      if (tile < 48) {
        const int mtx = tile / 6, ntx = tile - mtx * 6;
        v8f xac;
#pragma unroll
        for (int e = 0; e < 8; ++e) xac[e] = 0.0f;
#pragma unroll 1
        for (int kt = 0; kt < 8; ++kt) {
          v16h a  = *(const v16h*)(hNxt + (((mtx << 3) + kt) * 32 + lane) * 16);
          v16h bf = *(const v16h*)(bsS + ((kt * 6 + ntx) * 32 + lane) * 16);
          xac = __builtin_amdgcn_wmma_f32_16x16x32_f16(
              false, a, false, bf, (short)0, xac, false, false);
        }
#pragma unroll
        for (int r = 0; r < 8; ++r) {
          const int bi = (mtx << 4) + r + mrow_off;
          const int n  = (ntx << 4) + ncol;
          if (n < 88) xiL[bi * 96 + n] = xac[r] + b_xi[n];
        }
      }
    }
    __syncthreads();

    // ===== phase 3: DNC memory update (1 thread per batch element) =====
    if (tid < BB) {
      float* st = state + tid * 384;
      const float* xv = xiL + tid * 96;

      float rk[WCC], wkv[WCC], evv[WCC], wvv[WCC];
#pragma unroll
      for (int w = 0; w < WCC; ++w) rk[w]  = tanhf(xv[w]);
      float rs_ = softplusf(xv[20]);
#pragma unroll
      for (int w = 0; w < WCC; ++w) wkv[w] = tanhf(xv[21 + w]);
      float wstr = softplusf(xv[41]);
#pragma unroll
      for (int w = 0; w < WCC; ++w) evv[w] = sigm(xv[42 + w]);
#pragma unroll
      for (int w = 0; w < WCC; ++w) wvv[w] = tanhf(xv[62 + w]);
      float fg_ = sigm(xv[82]);
      float ag_ = sigm(xv[83]);
      float wg_ = sigm(xv[84]);
      float m3 = fmaxf(xv[85], fmaxf(xv[86], xv[87]));
      float e0 = __expf(xv[85] - m3), e1 = __expf(xv[86] - m3), e2 = __expf(xv[87] - m3);
      float ei = 1.0f / (e0 + e1 + e2);
      float rm_b = e0 * ei, rm_f = e1 * ei, rm_c = e2 * ei;

      float memv[MM][WCC], lk[MM][MM], prec[MM], rwv[MM], wwv[MM], us[MM];
      for (int m = 0; m < MM; ++m)
        for (int w = 0; w < WCC; ++w) memv[m][w] = st[m * WCC + w];
      for (int i = 0; i < MM; ++i)
        for (int j = 0; j < MM; ++j) lk[i][j] = st[200 + i * MM + j];
      for (int m = 0; m < MM; ++m) {
        prec[m] = st[300 + m]; rwv[m] = st[310 + m];
        wwv[m]  = st[320 + m]; us[m]  = st[330 + m];
      }

      // usage (uses previous ww, rw)
      for (int m = 0; m < MM; ++m) {
        float u = us[m] + (1.0f - us[m]) * wwv[m];
        us[m] = u * (1.0f - fg_ * rwv[m]);
      }
      // content-based write weights
      float kn = 0.0f;
      for (int w = 0; w < WCC; ++w) kn += wkv[w] * wkv[w];
      kn = sqrtf(kn) + EPSF;
      float dsc[MM], dmax = -1e30f;
      for (int m = 0; m < MM; ++m) {
        float nm = 0.0f, dot = 0.0f;
        for (int w = 0; w < WCC; ++w) { nm += memv[m][w] * memv[m][w]; dot += wkv[w] * memv[m][w]; }
        nm = sqrtf(nm) + EPSF;
        float v = (dot / (kn * nm)) * wstr;
        dsc[m] = v; dmax = fmaxf(dmax, v);
      }
      float cw[MM], ssum = 0.0f;
      for (int m = 0; m < MM; ++m) { cw[m] = __expf(dsc[m] - dmax); ssum += cw[m]; }
      float sinv = 1.0f / ssum;
      for (int m = 0; m < MM; ++m) cw[m] *= sinv;
      // allocation weights via stable argsort of usage
      float u2[MM]; int phi[MM];
      for (int m = 0; m < MM; ++m) { u2[m] = EPSF + (1.0f - EPSF) * us[m]; phi[m] = m; }
      for (int i = 1; i < MM; ++i) {
        int key = phi[i]; float kv = u2[key]; int j = i - 1;
        while (j >= 0 && u2[phi[j]] > kv) { phi[j + 1] = phi[j]; --j; }
        phi[j + 1] = key;
      }
      float alloc[MM], cp = 1.0f;
      for (int i = 0; i < MM; ++i) { float su = u2[phi[i]]; alloc[phi[i]] = (1.0f - su) * cp; cp *= su; }
      float wwn[MM];
      for (int m = 0; m < MM; ++m) wwn[m] = wg_ * (ag_ * alloc[m] + (1.0f - ag_) * cw[m]);
      // erase + write memory
      for (int m = 0; m < MM; ++m)
        for (int w = 0; w < WCC; ++w)
          memv[m][w] = memv[m][w] * (1.0f - wwn[m] * evv[w]) + wwn[m] * wvv[w];
      // temporal link + precedence
      float lkn[MM][MM];
      for (int i = 0; i < MM; ++i)
        for (int j = 0; j < MM; ++j)
          lkn[i][j] = (i == j) ? 0.0f
                               : ((1.0f - wwn[i] - wwn[j]) * lk[i][j] + wwn[i] * prec[j]);
      float wsum = 0.0f;
      for (int m = 0; m < MM; ++m) wsum += wwn[m];
      for (int m = 0; m < MM; ++m) prec[m] = (1.0f - wsum) * prec[m] + wwn[m];
      // read weights (content on NEW mem + temporal on OLD rw)
      float rkn = 0.0f;
      for (int w = 0; w < WCC; ++w) rkn += rk[w] * rk[w];
      rkn = sqrtf(rkn) + EPSF;
      float d2[MM]; dmax = -1e30f;
      for (int m = 0; m < MM; ++m) {
        float nm = 0.0f, dot = 0.0f;
        for (int w = 0; w < WCC; ++w) { nm += memv[m][w] * memv[m][w]; dot += rk[w] * memv[m][w]; }
        nm = sqrtf(nm) + EPSF;
        float v = (dot / (rkn * nm)) * rs_;
        d2[m] = v; dmax = fmaxf(dmax, v);
      }
      float cwr[MM]; ssum = 0.0f;
      for (int m = 0; m < MM; ++m) { cwr[m] = __expf(d2[m] - dmax); ssum += cwr[m]; }
      sinv = 1.0f / ssum;
      for (int m = 0; m < MM; ++m) cwr[m] *= sinv;
      float rwn[MM];
      for (int n = 0; n < MM; ++n) {
        float fwd = 0.0f, bwd = 0.0f;
        for (int m = 0; m < MM; ++m) { fwd += rwv[m] * lkn[n][m]; bwd += rwv[m] * lkn[m][n]; }
        rwn[n] = rm_b * bwd + rm_f * fwd + rm_c * cwr[n];
      }
      // store state
      for (int m = 0; m < MM; ++m)
        for (int w = 0; w < WCC; ++w) st[m * WCC + w] = memv[m][w];
      for (int i = 0; i < MM; ++i)
        for (int j = 0; j < MM; ++j) st[200 + i * MM + j] = lkn[i][j];
      for (int m = 0; m < MM; ++m) {
        st[300 + m] = prec[m]; st[310 + m] = rwn[m];
        st[320 + m] = wwn[m];  st[330 + m] = us[m];
      }
    }
    // next step's first barrier orders phase-3 reads vs. phase-2 xi writes
  }
  __syncthreads();

  // ===== final head: out = h_T @ W_fc.T + b_fc (final h lives in buffer 0) =====
#pragma unroll 1
  for (int rep = 0; rep < 2; ++rep) {
    const int tile = wid + (rep << 5);               // 0..63
    const int mtf = tile >> 3, ntf = tile & 7;
    v8f fac;
#pragma unroll
    for (int e = 0; e < 8; ++e) fac[e] = 0.0f;
#pragma unroll 1
    for (int kt = 0; kt < 8; ++kt) {
      v16h a  = *(const v16h*)(hA + (((mtf << 3) + kt) * 32 + lane) * 16);
      v16h bf = *(const v16h*)(wfc_p + (((kt << 3) + ntf) * 32 + lane) * 16);
      fac = __builtin_amdgcn_wmma_f32_16x16x32_f16(
          false, a, false, bf, (short)0, fac, false, false);
    }
#pragma unroll
    for (int r = 0; r < 8; ++r) {
      const int bi = (mtf << 4) + r + mrow_off;
      const int n  = (ntf << 4) + ncol;
      out[bi * OUTN + n] = fac[r] + b_fc[n];
    }
  }
}

// ---------------------------------------------------------------------------
extern "C" void kernel_launch(void* const* d_in, const int* in_sizes, int n_in,
                              void* d_out, int out_size, void* d_ws, size_t ws_size,
                              hipStream_t stream) {
  (void)in_sizes; (void)n_in; (void)out_size; (void)ws_size;
  const float* x    = (const float*)d_in[0];
  // d_in[1] input_lengths: unused by the reference
  const float* W_ih = (const float*)d_in[2];
  const float* W_hh = (const float*)d_in[3];
  const float* b_ih = (const float*)d_in[4];
  const float* b_hh = (const float*)d_in[5];
  const float* W_xi = (const float*)d_in[6];
  const float* b_xi = (const float*)d_in[7];
  const float* W_fc = (const float*)d_in[8];
  const float* b_fc = (const float*)d_in[9];
  float* out = (float*)d_out;

  char* ws = (char*)d_ws;
  _Float16* whh_p = (_Float16*)(ws + OFF_WHH);
  _Float16* wxi_p = (_Float16*)(ws + OFF_WXI);
  _Float16* wfc_p = (_Float16*)(ws + OFF_WFC);
  float*    state = (float*)(ws + OFF_STATE);
  _Float16* gxp   = (_Float16*)(ws + OFF_GX);

  pack_whh_kernel<<<262144 / 256, 256, 0, stream>>>(W_hh, whh_p);
  pack_wxi_kernel<<<24576 / 256, 256, 0, stream>>>(W_xi, wxi_p);
  pack_wfc_kernel<<<32768 / 256, 256, 0, stream>>>(W_fc, wfc_p);
  gx_kernel<<<262144, 256, 0, stream>>>(x, W_ih, b_ih, b_hh, gxp);

  dnc_persistent_kernel<<<1, 1024, LDS_BYTES, stream>>>(
      whh_p, wxi_p, wfc_p, gxp, state, b_xi, b_fc, out);
}
